// GNNBlock_31233002177084
// MI455X (gfx1250) — matrix operand order repaired and checked by reading, not compiled, not run on hardware
//
#include <hip/hip_runtime.h>
#include <hip/hip_bf16.h>
#include <math.h>

// ---------------- problem constants ----------------
#define N_NODES 2048
#define DMODEL  256
#define BATCH   4
#define NHEAD   4
#define HDIM    64
#define HIDDEN  512
#define BN      (BATCH * N_NODES)
#define POW_ITERS 40

// ---------------- CDNA5 WMMA types ----------------
typedef __bf16 bf16_t;
typedef bf16_t bfx16 __attribute__((ext_vector_type(16)));
typedef float  fx8   __attribute__((ext_vector_type(8)));
typedef unsigned short usx8 __attribute__((ext_vector_type(8)));

union BF16Frag { bfx16 v; usx8 u[2]; };

#define WMMA_BF16(Af, Bf, Cacc) \
  __builtin_amdgcn_wmma_f32_16x16x32_bf16(false, (Af).v, false, (Bf).v, (short)0, (Cacc), false, false)

__device__ __forceinline__ unsigned short f2bf(float f) {
  union { float f; unsigned u; } v; v.f = f;
  unsigned r = v.u + 0x7FFFu + ((v.u >> 16) & 1u);   // round-to-nearest-even
  return (unsigned short)(r >> 16);
}

// ---------------- power iteration for lambda_max ----------------
__global__ void pow_init(float* v, float* scal) {
  int i = blockIdx.x * blockDim.x + threadIdx.x;
  if (i < N_NODES) {
    unsigned u = (unsigned)i * 2654435761u;
    v[i] = (float)((u >> 16) & 0x7FFF) / 32768.0f + 0.01f;  // hash seed (NOT ones: L@1 == 0)
  }
  if (i == 0) { scal[0] = 0.f; scal[1] = 0.f; scal[2] = 0.f; scal[3] = 0.f; }
}

__global__ void pow_matvec(const float* __restrict__ L, const float* __restrict__ v,
                           float* __restrict__ w, float* __restrict__ ss) {
  int row = blockIdx.x * blockDim.x + threadIdx.x;
  const float* Lr = L + (size_t)row * N_NODES;
  float sum = 0.f;
  for (int k = 0; k < N_NODES; ++k) sum += Lr[k] * v[k];
  w[row] = sum;
  __shared__ float sm[256];
  sm[threadIdx.x] = sum * sum;
  __syncthreads();
  for (int s = 128; s > 0; s >>= 1) {
    if (threadIdx.x < s) sm[threadIdx.x] += sm[threadIdx.x + s];
    __syncthreads();
  }
  if (threadIdx.x == 0) atomicAdd(ss, sm[0]);
}

__global__ void pow_scale(const float* __restrict__ w, float* __restrict__ v,
                          const float* __restrict__ ss, float* __restrict__ ss_next) {
  int i = blockIdx.x * blockDim.x + threadIdx.x;
  float inv = rsqrtf(*ss + 1e-30f);
  v[i] = w[i] * inv;
  if (i == 0) *ss_next = 0.f;           // zero slot for next iteration
}

__global__ void pow_finalize(const float* __restrict__ ss, float* __restrict__ inv2) {
  *inv2 = 2.0f / (sqrtf(*ss) + 1e-8f);  // 2 / lam_max   (ref adds 1e-8)
}

// Lb = bf16(2*L/lam - I)   -> folds the whole LaplacianConv normalization
__global__ void build_lb(const float* __restrict__ L, const float* __restrict__ inv2p,
                         unsigned short* __restrict__ Lb) {
  size_t idx = (size_t)blockIdx.x * 256 + threadIdx.x;
  int i = (int)(idx / N_NODES), j = (int)(idx % N_NODES);
  float val = L[idx] * (*inv2p) - (i == j ? 1.0f : 0.0f);
  Lb[idx] = f2bf(val);
}

// ---------------- generic converts ----------------
__global__ void convert_bf16(const float* __restrict__ in, unsigned short* __restrict__ out, size_t n) {
  size_t i = (size_t)blockIdx.x * 256 + threadIdx.x;
  if (i < n) out[i] = f2bf(in[i]);
}

// out[c*R + r] = bf16(in[r*C + c]); batched via blockIdx.z
__global__ void transpose_to_bf16(const float* __restrict__ in, unsigned short* __restrict__ out,
                                  int R, int C, long long sIn, long long sOut) {
  const float* ib = in + (size_t)blockIdx.z * sIn;
  unsigned short* ob = out + (size_t)blockIdx.z * sOut;
  size_t idx = (size_t)blockIdx.x * 256 + threadIdx.x;
  if (idx < (size_t)R * C) {
    int r = (int)(idx / C), c = (int)(idx % C);
    ob[(size_t)c * R + r] = f2bf(ib[(size_t)r * C + c]);
  }
}

// ---------------- WMMA GEMM: C[M,Nn] = A[M,K] @ B[K,Nn] (+bias) ----------------
// A: bf16 row-major (lda). BT: bf16, B transposed, row-major (ldbt = K stride).
// One wave computes a 64x64 tile; 16 v_wmma_f32_16x16x32_bf16 per K-step of 32.
__global__ void __launch_bounds__(32) gemm_wmma_bf16(
    const unsigned short* __restrict__ A, const unsigned short* __restrict__ BT,
    const float* __restrict__ bias, float* __restrict__ Cf, unsigned short* __restrict__ Cb,
    int M, int Nn, int Kk, int lda, int ldbt, int ldc,
    long long sA, long long sBT, long long sC) {
  const int bz = blockIdx.z;
  A  += (size_t)bz * sA;
  BT += (size_t)bz * sBT;
  const int n0 = blockIdx.x * 64, m0 = blockIdx.y * 64;
  const int lane = threadIdx.x, lh = lane >> 4, ll = lane & 15;
  const fx8 z = {0, 0, 0, 0, 0, 0, 0, 0};
  fx8 acc[4][4];
#pragma unroll
  for (int i = 0; i < 4; ++i)
#pragma unroll
    for (int j = 0; j < 4; ++j) acc[i][j] = z;

  for (int k0 = 0; k0 < Kk; k0 += 32) {
    BF16Frag af[4], bfr[4];
#pragma unroll
    for (int i = 0; i < 4; ++i) {   // A frag: lane row = m0+16i+ll, k = k0+lh*16 .. +15 contiguous
      const unsigned short* p = A + (size_t)(m0 + 16 * i + ll) * lda + k0 + lh * 16;
      af[i].u[0] = *(const usx8*)p;
      af[i].u[1] = *(const usx8*)(p + 8);
    }
#pragma unroll
    for (int j = 0; j < 4; ++j) {   // B frag: lane col = n0+16j+ll, k = k0+lh*16 .. +15 contiguous in BT
      const unsigned short* p = BT + (size_t)(n0 + 16 * j + ll) * ldbt + k0 + lh * 16;
      bfr[j].u[0] = *(const usx8*)p;
      bfr[j].u[1] = *(const usx8*)(p + 8);
    }
#pragma unroll
    for (int i = 0; i < 4; ++i)
#pragma unroll
      for (int j = 0; j < 4; ++j) acc[i][j] = WMMA_BF16(af[i], bfr[j], acc[i][j]);
  }

  float* Cfb = Cf ? Cf + (size_t)bz * sC : nullptr;
  unsigned short* Cbb = Cb ? Cb + (size_t)bz * sC : nullptr;
#pragma unroll
  for (int j = 0; j < 4; ++j) {
    const int col = n0 + 16 * j + ll;
    const float bv = bias ? bias[col] : 0.0f;
#pragma unroll
    for (int i = 0; i < 4; ++i) {
#pragma unroll
      for (int r = 0; r < 8; ++r) {     // C layout: lane l, vgpr r -> row r + 8*(l/16), col l%16
        const size_t row = (size_t)(m0 + 16 * i + r + 8 * lh);
        const float vv = acc[i][j][r] + bv;
        if (Cfb) Cfb[row * ldc + col] = vv;
        if (Cbb) Cbb[row * ldc + col] = f2bf(vv);
      }
    }
  }
}

// ---------------- flash-style adjacency-masked attention ----------------
// One wave handles 16 query rows of one (b,h). Streams 32 key columns per step.
// Qb/Kb: bf16 (B*N, D) row-major.  Vt: bf16 per-batch transposed (D, N).  O: f32 (B,N,D).
__global__ void __launch_bounds__(32) flash_attn_kernel(
    const unsigned short* __restrict__ Qb, const unsigned short* __restrict__ Kb,
    const unsigned short* __restrict__ Vt, const float* __restrict__ adj,
    float* __restrict__ O) {
  __shared__ __align__(16) unsigned short plds[16 * 32];
  const int i0 = blockIdx.x * 16, h = blockIdx.y, b = blockIdx.z;
  const int lane = threadIdx.x, lh = lane >> 4, ll = lane & 15;
  const fx8 z = {0, 0, 0, 0, 0, 0, 0, 0};

  BF16Frag aq0, aq1;     // Q tile 16x64 as two K=32 A-fragments
  {
    const unsigned short* p = Qb + (size_t)(b * N_NODES + i0 + ll) * DMODEL + h * HDIM;
    aq0.u[0] = *(const usx8*)(p + lh * 16);
    aq0.u[1] = *(const usx8*)(p + 8 + lh * 16);
    aq1.u[0] = *(const usx8*)(p + 32 + lh * 16);
    aq1.u[1] = *(const usx8*)(p + 40 + lh * 16);
  }

  float mrow[8], lsum[8];
  fx8 accO[4];
#pragma unroll
  for (int r = 0; r < 8; ++r) { mrow[r] = -INFINITY; lsum[r] = 0.0f; }
#pragma unroll
  for (int t4 = 0; t4 < 4; ++t4) accO[t4] = z;

  for (int j0 = 0; j0 < N_NODES; j0 += 32) {
    fx8 s0 = z, s1 = z;  // scores for cols j0..j0+15 and j0+16..j0+31
    {
      BF16Frag bk;       // B = K^T: lane col = key row, k = head-dim (contiguous in Kb)
      const unsigned short* p = Kb + (size_t)(b * N_NODES + j0 + ll) * DMODEL + h * HDIM + lh * 16;
      bk.u[0] = *(const usx8*)p;        bk.u[1] = *(const usx8*)(p + 8);
      s0 = WMMA_BF16(aq0, bk, s0);
      bk.u[0] = *(const usx8*)(p + 32); bk.u[1] = *(const usx8*)(p + 40);
      s0 = WMMA_BF16(aq1, bk, s0);
      const unsigned short* q = p + 16 * DMODEL;
      bk.u[0] = *(const usx8*)q;        bk.u[1] = *(const usx8*)(q + 8);
      s1 = WMMA_BF16(aq0, bk, s1);
      bk.u[0] = *(const usx8*)(q + 32); bk.u[1] = *(const usx8*)(q + 40);
      s1 = WMMA_BF16(aq1, bk, s1);
    }
#pragma unroll
    for (int r = 0; r < 8; ++r) {                 // C-layout row = i0 + r + 8*lh, col = j0(+16) + ll
      const int gm = i0 + r + 8 * lh;
      const float a0 = adj[(size_t)gm * N_NODES + j0 + ll];
      const float a1 = adj[(size_t)gm * N_NODES + j0 + 16 + ll];
      float v0 = (a0 > 0.0f) ? s0[r] * 0.125f : -1e9f;   // 1/sqrt(64)
      float v1 = (a1 > 0.0f) ? s1[r] * 0.125f : -1e9f;
      float t = fmaxf(v0, v1);
#pragma unroll
      for (int off = 1; off < 16; off <<= 1) t = fmaxf(t, __shfl_xor(t, off, 32));
      const float nm = fmaxf(mrow[r], t);
      const float al = __expf(mrow[r] - nm);
      mrow[r] = nm;
      const float p0 = __expf(v0 - nm), p1 = __expf(v1 - nm);
      float rs = p0 + p1;
#pragma unroll
      for (int off = 1; off < 16; off <<= 1) rs += __shfl_xor(rs, off, 32);
      lsum[r] = lsum[r] * al + rs;
      plds[(r + 8 * lh) * 32 + ll]      = f2bf(p0);   // stage P in LDS (C-layout -> A-layout hop)
      plds[(r + 8 * lh) * 32 + 16 + ll] = f2bf(p1);
#pragma unroll
      for (int t4 = 0; t4 < 4; ++t4) accO[t4][r] *= al;
    }
    __syncthreads();
    BF16Frag ap;                                   // reload P as A-fragment (16x32, K=32)
    ap.u[0] = *(const usx8*)(plds + ll * 32 + lh * 16);
    ap.u[1] = *(const usx8*)(plds + ll * 32 + lh * 16 + 8);
#pragma unroll
    for (int t4 = 0; t4 < 4; ++t4) {
      BF16Frag bv;                                 // B = V tile 32x16: contiguous along nodes in Vt
      const unsigned short* p = Vt + ((size_t)b * DMODEL + h * HDIM + t4 * 16 + ll) * N_NODES + j0 + lh * 16;
      bv.u[0] = *(const usx8*)p;
      bv.u[1] = *(const usx8*)(p + 8);
      accO[t4] = WMMA_BF16(ap, bv, accO[t4]);
    }
    __syncthreads();
  }
#pragma unroll
  for (int r = 0; r < 8; ++r) {
    const float inv = 1.0f / lsum[r];
    const size_t row = (size_t)(b * N_NODES + i0 + r + 8 * lh);
#pragma unroll
    for (int t4 = 0; t4 < 4; ++t4)
      O[row * DMODEL + h * HDIM + t4 * 16 + ll] = accO[t4][r] * inv;
  }
}

// ---------------- layernorm fusion (one block per row, D=256 threads) ----------------
// mode 0: y = relu(LN(a)) + b      (b = residual input x)
// mode 1: y = LN(a + b)
__global__ void ln_fuse(const float* __restrict__ a, const float* __restrict__ b,
                        const float* __restrict__ gamma, const float* __restrict__ beta,
                        float* __restrict__ outf, unsigned short* __restrict__ outb, int mode) {
  const int row = blockIdx.x, c = threadIdx.x;
  const size_t idx = (size_t)row * DMODEL + c;
  float t = (mode == 0) ? a[idx] : (a[idx] + b[idx]);
  __shared__ float sm[DMODEL];
  sm[c] = t; __syncthreads();
  for (int s = DMODEL / 2; s > 0; s >>= 1) { if (c < s) sm[c] += sm[c + s]; __syncthreads(); }
  const float mu = sm[0] * (1.0f / DMODEL);
  __syncthreads();
  const float d = t - mu;
  sm[c] = d * d; __syncthreads();
  for (int s = DMODEL / 2; s > 0; s >>= 1) { if (c < s) sm[c] += sm[c + s]; __syncthreads(); }
  const float var = sm[0] * (1.0f / DMODEL);
  float y = d * rsqrtf(var + 1e-5f) * gamma[c] + beta[c];
  if (mode == 0) y = fmaxf(y, 0.0f) + b[idx];
  if (outf) outf[idx] = y;
  if (outb) outb[idx] = f2bf(y);
}

// exact GELU -> bf16
__global__ void gelu_bf16(const float* __restrict__ in, unsigned short* __restrict__ out, size_t n) {
  size_t i = (size_t)blockIdx.x * 256 + threadIdx.x;
  if (i < n) {
    float x = in[i];
    out[i] = f2bf(0.5f * x * (1.0f + erff(x * 0.70710678118654752f)));
  }
}

// ---------------- host orchestration ----------------
extern "C" void kernel_launch(void* const* d_in, const int* in_sizes, int n_in,
                              void* d_out, int out_size, void* d_ws, size_t ws_size,
                              hipStream_t stream) {
  (void)in_sizes; (void)n_in; (void)out_size; (void)ws_size;
  const float* x   = (const float*)d_in[0];
  const float* L   = (const float*)d_in[1];
  const float* adj = (const float*)d_in[2];
  const float* Wc  = (const float*)d_in[3];
  const float* bc  = (const float*)d_in[4];
  const float* g1  = (const float*)d_in[5];
  const float* be1 = (const float*)d_in[6];
  const float* Wq  = (const float*)d_in[7];
  const float* Wk  = (const float*)d_in[8];
  const float* Wv  = (const float*)d_in[9];
  const float* Wo  = (const float*)d_in[10];
  const float* bo  = (const float*)d_in[11];
  const float* g2  = (const float*)d_in[12];
  const float* be2 = (const float*)d_in[13];
  const float* W1  = (const float*)d_in[14];
  const float* b1  = (const float*)d_in[15];
  const float* W2  = (const float*)d_in[16];
  const float* b2  = (const float*)d_in[17];
  const float* g3  = (const float*)d_in[18];
  const float* be3 = (const float*)d_in[19];
  float* out = (float*)d_out;

  char* ws = (char*)d_ws;
  size_t off = 0;
  auto carve = [&](size_t bytes) -> char* {
    char* p = ws + off;
    off = (off + bytes + 255) & ~(size_t)255;
    return p;
  };
  float*          SCAL = (float*)carve(256);                                   // [0/1]=ss slots [2]=2/lam
  float*          V0   = (float*)carve(sizeof(float) * N_NODES);
  float*          W0   = (float*)carve(sizeof(float) * N_NODES);
  unsigned short* LB   = (unsigned short*)carve(2ull * N_NODES * N_NODES);
  unsigned short* XT   = (unsigned short*)carve(2ull * BATCH * DMODEL * N_NODES);
  unsigned short* XPB  = (unsigned short*)carve(2ull * BN * DMODEL);
  float*          BUFA = (float*)carve(4ull * BN * DMODEL);                    // xp -> Vf32 -> O -> o2 -> m2
  float*          X1F  = (float*)carve(4ull * BN * DMODEL);
  unsigned short* X1B  = (unsigned short*)carve(2ull * BN * DMODEL);
  unsigned short* QB   = (unsigned short*)carve(2ull * BN * DMODEL);
  unsigned short* KB   = (unsigned short*)carve(2ull * BN * DMODEL);
  unsigned short* VT   = (unsigned short*)carve(2ull * BATCH * DMODEL * N_NODES);
  unsigned short* OB   = (unsigned short*)carve(2ull * BN * DMODEL);
  float*          X2F  = (float*)carve(4ull * BN * DMODEL);
  unsigned short* X2B  = (unsigned short*)carve(2ull * BN * DMODEL);
  float*          M1F  = (float*)carve(4ull * BN * HIDDEN);
  unsigned short* M1G  = (unsigned short*)carve(2ull * BN * HIDDEN);
  unsigned short* WCT  = (unsigned short*)carve(2ull * DMODEL * DMODEL);
  unsigned short* WQT  = (unsigned short*)carve(2ull * DMODEL * DMODEL);
  unsigned short* WKT  = (unsigned short*)carve(2ull * DMODEL * DMODEL);
  unsigned short* WVT  = (unsigned short*)carve(2ull * DMODEL * DMODEL);
  unsigned short* WOT  = (unsigned short*)carve(2ull * DMODEL * DMODEL);
  unsigned short* W1T  = (unsigned short*)carve(2ull * HIDDEN * DMODEL);
  unsigned short* W2T  = (unsigned short*)carve(2ull * DMODEL * HIDDEN);

  const long long sXT = (long long)DMODEL * N_NODES;
  const long long sX  = (long long)N_NODES * DMODEL;

  // --- weight prep (bf16, transposed so GEMM B-fragments are contiguous) ---
  transpose_to_bf16<<<dim3((DMODEL * DMODEL) / 256, 1, 1), 256, 0, stream>>>(Wc, WCT, DMODEL, DMODEL, 0, 0);
  transpose_to_bf16<<<dim3((DMODEL * DMODEL) / 256, 1, 1), 256, 0, stream>>>(Wq, WQT, DMODEL, DMODEL, 0, 0);
  transpose_to_bf16<<<dim3((DMODEL * DMODEL) / 256, 1, 1), 256, 0, stream>>>(Wk, WKT, DMODEL, DMODEL, 0, 0);
  transpose_to_bf16<<<dim3((DMODEL * DMODEL) / 256, 1, 1), 256, 0, stream>>>(Wv, WVT, DMODEL, DMODEL, 0, 0);
  transpose_to_bf16<<<dim3((DMODEL * DMODEL) / 256, 1, 1), 256, 0, stream>>>(Wo, WOT, DMODEL, DMODEL, 0, 0);
  transpose_to_bf16<<<dim3((DMODEL * HIDDEN) / 256, 1, 1), 256, 0, stream>>>(W1, W1T, DMODEL, HIDDEN, 0, 0);
  transpose_to_bf16<<<dim3((HIDDEN * DMODEL) / 256, 1, 1), 256, 0, stream>>>(W2, W2T, HIDDEN, DMODEL, 0, 0);

  // --- lambda_max via power iteration (ping-pong norm slots) ---
  pow_init<<<N_NODES / 256, 256, 0, stream>>>(V0, SCAL);
  for (int it = 0; it < POW_ITERS; ++it) {
    pow_matvec<<<N_NODES / 256, 256, 0, stream>>>(L, V0, W0, SCAL + (it & 1));
    pow_scale<<<N_NODES / 256, 256, 0, stream>>>(W0, V0, SCAL + (it & 1), SCAL + ((it + 1) & 1));
  }
  pow_matvec<<<N_NODES / 256, 256, 0, stream>>>(L, V0, W0, SCAL + (POW_ITERS & 1));
  pow_finalize<<<1, 1, 0, stream>>>(SCAL + (POW_ITERS & 1), SCAL + 2);

  // --- Lb = bf16(2L/lam - I); Xt = bf16(x^T) per batch ---
  build_lb<<<(N_NODES * N_NODES) / 256, 256, 0, stream>>>(L, SCAL + 2, LB);
  transpose_to_bf16<<<dim3((N_NODES * DMODEL) / 256, 1, BATCH), 256, 0, stream>>>(x, XT, N_NODES, DMODEL, sX, sXT);

  // --- xp = L_norm @ x  (batched WMMA GEMM, A shared across batch) ---
  gemm_wmma_bf16<<<dim3(DMODEL / 64, N_NODES / 64, BATCH), 32, 0, stream>>>(
      LB, XT, nullptr, BUFA, nullptr, N_NODES, DMODEL, N_NODES,
      N_NODES, N_NODES, DMODEL, 0, sXT, sX);
  convert_bf16<<<(BN * DMODEL) / 256, 256, 0, stream>>>(BUFA, XPB, (size_t)BN * DMODEL);

  // --- h = xp @ Wc + bc ; x1 = relu(LN(h)) + x ---
  gemm_wmma_bf16<<<dim3(DMODEL / 64, BN / 64, 1), 32, 0, stream>>>(
      XPB, WCT, bc, BUFA, nullptr, BN, DMODEL, DMODEL, DMODEL, DMODEL, DMODEL, 0, 0, 0);
  ln_fuse<<<BN, DMODEL, 0, stream>>>(BUFA, x, g1, be1, X1F, X1B, 0);

  // --- Q, K (bf16 direct), V (f32 then transpose for PV B-fragments) ---
  gemm_wmma_bf16<<<dim3(DMODEL / 64, BN / 64, 1), 32, 0, stream>>>(
      X1B, WQT, nullptr, nullptr, QB, BN, DMODEL, DMODEL, DMODEL, DMODEL, DMODEL, 0, 0, 0);
  gemm_wmma_bf16<<<dim3(DMODEL / 64, BN / 64, 1), 32, 0, stream>>>(
      X1B, WKT, nullptr, nullptr, KB, BN, DMODEL, DMODEL, DMODEL, DMODEL, DMODEL, 0, 0, 0);
  gemm_wmma_bf16<<<dim3(DMODEL / 64, BN / 64, 1), 32, 0, stream>>>(
      X1B, WVT, nullptr, BUFA, nullptr, BN, DMODEL, DMODEL, DMODEL, DMODEL, DMODEL, 0, 0, 0);
  transpose_to_bf16<<<dim3((N_NODES * DMODEL) / 256, 1, BATCH), 256, 0, stream>>>(BUFA, VT, N_NODES, DMODEL, sX, sXT);

  // --- adjacency-masked flash attention -> O (f32, reuses BUFA) ---
  flash_attn_kernel<<<dim3(N_NODES / 16, NHEAD, BATCH), 32, 0, stream>>>(QB, KB, VT, adj, BUFA);
  convert_bf16<<<(BN * DMODEL) / 256, 256, 0, stream>>>(BUFA, OB, (size_t)BN * DMODEL);

  // --- o2 = O @ Wo + bo ; x2 = LN(x1 + o2) ---
  gemm_wmma_bf16<<<dim3(DMODEL / 64, BN / 64, 1), 32, 0, stream>>>(
      OB, WOT, bo, BUFA, nullptr, BN, DMODEL, DMODEL, DMODEL, DMODEL, DMODEL, 0, 0, 0);
  ln_fuse<<<BN, DMODEL, 0, stream>>>(X1F, BUFA, g2, be2, X2F, X2B, 1);

  // --- MLP: m = gelu(x2 @ W1 + b1) @ W2 + b2 ; out = LN(x2 + m) ---
  gemm_wmma_bf16<<<dim3(HIDDEN / 64, BN / 64, 1), 32, 0, stream>>>(
      X2B, W1T, b1, M1F, nullptr, BN, HIDDEN, DMODEL, DMODEL, DMODEL, HIDDEN, 0, 0, 0);
  gelu_bf16<<<(BN * HIDDEN) / 256, 256, 0, stream>>>(M1F, M1G, (size_t)BN * HIDDEN);
  gemm_wmma_bf16<<<dim3(DMODEL / 64, BN / 64, 1), 32, 0, stream>>>(
      M1G, W2T, b2, BUFA, nullptr, BN, DMODEL, HIDDEN, HIDDEN, HIDDEN, DMODEL, 0, 0, 0);
  ln_fuse<<<BN, DMODEL, 0, stream>>>(X2F, BUFA, g3, be3, out, nullptr, 1);
}
